// Edge_cycle_50869592655484
// MI455X (gfx1250) — matrix operand order repaired and checked by reading, not compiled, not run on hardware
//
#include <hip/hip_runtime.h>

// ---------------- problem constants ----------------
#define DD   64
#define HH   128
#define NEg  200000
#define R5g  150000
#define R6g  180000
#define RTg  330000      // R5+R6
#define NC5g 30000
#define NC6g 30000

typedef __attribute__((ext_vector_type(16))) __bf16 v16bf;
typedef __attribute__((ext_vector_type(8)))  float  v8f;

#define WAVE_FENCE() asm volatile("" ::: "memory")

__device__ __forceinline__ v8f wmma_bf16(v16bf a, v16bf b, v8f c) {
  // D = A(16x32 bf16) x B(32x16 bf16) + C(16x16 f32)
  return __builtin_amdgcn_wmma_f32_16x16x32_bf16(
      /*neg_a=*/false, a, /*neg_b=*/false, b,
      /*c_mod=*/(short)0, c, /*reuse_a=*/false, /*reuse_b=*/false);
}

__device__ __forceinline__ v8f zero8() {
  v8f z;
#pragma unroll
  for (int i = 0; i < 8; ++i) z[i] = 0.f;
  return z;
}

// Build 16x32 bf16 A-fragment from an LDS tile (row-major, leading dim ld).
// ISA 7.12.2 (16-bit A): lane L holds row M=L%16; half=L/16 selects K-group:
//   VGPR j (j<4):  K = kbase + half*8 + 2j,2j+1
//   VGPR j (j>=4): K = kbase + half*8 + 16 + 2(j-4), +1
__device__ __forceinline__ v16bf load_a(const __bf16* lds, int ld, int kbase, int lane) {
  const int half = (lane >> 4) & 1;
  const int m = lane & 15;
  const __bf16* p = lds + m * ld + kbase + half * 8;
  union { v16bf v; unsigned u[8]; } a;
#pragma unroll
  for (int j = 0; j < 8; ++j) {
    const int k = (j < 4) ? (2 * j) : (16 + 2 * (j - 4));
    a.u[j] = *reinterpret_cast<const unsigned*>(p + k);
  }
  return a.v;
}

// ---------------- weight fragment packing ----------------
// Packed layout: fragment f=(kt*NNT+nt); lane's 16 bf16 contiguous at
// (f*32+lane)*16  -> one 32B load per lane in the GEMMs.
__global__ void pack_w(const float* __restrict__ W, __bf16* __restrict__ out,
                       int K, int N) {
  const int gid = blockIdx.x * blockDim.x + threadIdx.x;
  const int NNT = N >> 4, NKT = K >> 5;
  if (gid >= NKT * NNT * 32) return;
  const int lane = gid & 31, f = gid >> 5;
  const int kt = f / NNT, nt = f % NNT;
  const int col = nt * 16 + (lane & 15);
  const int kbase = kt * 32 + ((lane >> 4) & 1) * 8;  // B: lanes 16-31 hold K+8 (and K+24)
  __bf16* dst = out + ((size_t)f * 32 + lane) * 16;
#pragma unroll
  for (int j = 0; j < 8; ++j) {
    const int k = kbase + ((j < 4) ? (2 * j) : (16 + 2 * (j - 4)));
    dst[2 * j]     = (__bf16)W[(size_t)k * N + col];
    dst[2 * j + 1] = (__bf16)W[(size_t)(k + 1) * N + col];
  }
}

// ---------------- stage 1: cycle_mlp_2 ----------------
// x = relu(relu([cyc || e0+e1] @ W1 + b1) @ W2 + b2)  -> x_all
__global__ void k1_stage1(const float* __restrict__ edge,
                          const float* __restrict__ cyc5,
                          const float* __restrict__ cyc6,
                          const int* __restrict__ e2c5,
                          const int* __restrict__ e2c6,
                          const __bf16* __restrict__ W1p, const float* __restrict__ b1,
                          const __bf16* __restrict__ W2p, const float* __restrict__ b2,
                          float* __restrict__ x_all) {
  extern __shared__ char smem[];
  const int lane = threadIdx.x & 31;
  const int wid = threadIdx.x >> 5;
  const int tile = blockIdx.x * 4 + wid;
  if (tile >= RTg / 16) return;                 // wave-uniform: EXEC stays all-1s
  __bf16* ain = (__bf16*)(smem + wid * 8192);   // [16][128] bf16
  __bf16* h1 = ain + 16 * 128;                  // [16][128] bf16

  const int row0 = tile * 16;
  const float* cptr;
  const int* iptr;
  if (row0 < R5g) { cptr = cyc5 + (size_t)row0 * DD;        iptr = e2c5 + (size_t)row0 * 2; }
  else            { cptr = cyc6 + (size_t)(row0 - R5g) * DD; iptr = e2c6 + (size_t)(row0 - R5g) * 2; }

  for (int t = lane; t < 16 * DD; t += 32) {
    const int r = t >> 6, c = t & 63;
    ain[r * 128 + c] = (__bf16)cptr[r * DD + c];
    const int i0 = iptr[2 * r], i1 = iptr[2 * r + 1];
    ain[r * 128 + DD + c] = (__bf16)(edge[(size_t)i0 * DD + c] + edge[(size_t)i1 * DD + c]);
  }
  WAVE_FENCE();

  const int col = lane & 15, half = (lane >> 4) & 1;
  v16bf afr[4];
#pragma unroll
  for (int kt = 0; kt < 4; ++kt) afr[kt] = load_a(ain, 128, kt * 32, lane);
  const v16bf* bW1 = (const v16bf*)W1p;
  for (int nt = 0; nt < 8; ++nt) {
    v8f acc = zero8();
#pragma unroll
    for (int kt = 0; kt < 4; ++kt)
      acc = wmma_bf16(afr[kt], bW1[(kt * 8 + nt) * 32 + lane], acc);
    const float bias = b1[nt * 16 + col];
#pragma unroll
    for (int r = 0; r < 8; ++r) {
      const float v = fmaxf(acc[r] + bias, 0.f);
      h1[(r + 8 * half) * 128 + nt * 16 + col] = (__bf16)v;
    }
  }
  WAVE_FENCE();

  v16bf af2[4];
#pragma unroll
  for (int kt = 0; kt < 4; ++kt) af2[kt] = load_a(h1, 128, kt * 32, lane);
  const v16bf* bW2 = (const v16bf*)W2p;
  for (int nt = 0; nt < 4; ++nt) {
    v8f acc = zero8();
#pragma unroll
    for (int kt = 0; kt < 4; ++kt)
      acc = wmma_bf16(af2[kt], bW2[(kt * 4 + nt) * 32 + lane], acc);
    const float bias = b2[nt * 16 + col];
#pragma unroll
    for (int r = 0; r < 8; ++r)
      x_all[(size_t)(row0 + r + 8 * half) * DD + nt * 16 + col] = fmaxf(acc[r] + bias, 0.f);
  }
}

// ---------------- stage 1b: per-cycle segment sums ----------------
__global__ void k2_segsum(const float* __restrict__ x_all, float* __restrict__ ssum) {
  const int gid = blockIdx.x * blockDim.x + threadIdx.x;
  if (gid >= (NC5g + NC6g) * 16) return;
  const int c = gid >> 4, q = (gid & 15) * 4;
  const float* base;
  int s;
  if (c < NC5g) { base = x_all + (size_t)c * 5 * DD + q; s = 5; }
  else          { base = x_all + (size_t)R5g * DD + (size_t)(c - NC5g) * 6 * DD + q; s = 6; }
  float4 acc = {0.f, 0.f, 0.f, 0.f};
  for (int j = 0; j < s; ++j) {
    const float4 v = *reinterpret_cast<const float4*>(base + (size_t)j * DD);
    acc.x += v.x; acc.y += v.y; acc.z += v.z; acc.w += v.w;
  }
  *reinterpret_cast<float4*>(ssum + (size_t)c * DD + q) = acc;
}

// ---------------- stage 2: autobahn + cycle_mlp ----------------
__global__ void k3_stage2(const float* __restrict__ x_all, const float* __restrict__ ssum,
                          const __bf16* __restrict__ Widp, const __bf16* __restrict__ Wsump,
                          const float* __restrict__ bab,
                          const __bf16* __restrict__ Wc1p, const float* __restrict__ bc1,
                          const __bf16* __restrict__ Wc2p, const float* __restrict__ bc2,
                          const __bf16* __restrict__ Wc3p, const float* __restrict__ bc3,
                          float* __restrict__ out, int rows, int s, int row_off, int cyc_off) {
  extern __shared__ char smem[];
  const int lane = threadIdx.x & 31;
  const int wid = threadIdx.x >> 5;
  const int tile = blockIdx.x * 2 + wid;
  if (tile * 16 >= rows) return;
  __bf16* y = (__bf16*)(smem + wid * 16384);  // [16][384] bf16
  __bf16* h = y + 16 * 384;                   // [16][128] bf16

  const int row0 = tile * 16;
  for (int t = lane; t < 16 * DD; t += 32) {
    const int r = t >> 6, c = t & 63;
    const int rr = row0 + r;
    y[r * 384 + c]      = (__bf16)x_all[(size_t)(row_off + rr) * DD + c];
    y[r * 384 + DD + c] = (__bf16)ssum[(size_t)(cyc_off + rr / s) * DD + c];
  }
  WAVE_FENCE();

  const int col = lane & 15, half = (lane >> 4) & 1;

  // aut = x@Wid + bsum@Wsum + bab  -> y[:,128:384]
  v16bf ax0 = load_a(y, 384, 0, lane),  ax1 = load_a(y, 384, 32, lane);
  v16bf ab0 = load_a(y, 384, 64, lane), ab1 = load_a(y, 384, 96, lane);
  const v16bf* bid = (const v16bf*)Widp;
  const v16bf* bsm = (const v16bf*)Wsump;
  for (int nt = 0; nt < 16; ++nt) {
    v8f acc = zero8();
    acc = wmma_bf16(ax0, bid[(0 * 16 + nt) * 32 + lane], acc);
    acc = wmma_bf16(ax1, bid[(1 * 16 + nt) * 32 + lane], acc);
    acc = wmma_bf16(ab0, bsm[(0 * 16 + nt) * 32 + lane], acc);
    acc = wmma_bf16(ab1, bsm[(1 * 16 + nt) * 32 + lane], acc);
    const float bias = bab[nt * 16 + col];
#pragma unroll
    for (int r = 0; r < 8; ++r)
      y[(r + 8 * half) * 384 + 128 + nt * 16 + col] = (__bf16)(acc[r] + bias);
  }
  WAVE_FENCE();

  // h = relu(y @ Wc1 + bc1), K=384
  v16bf ay[12];
#pragma unroll
  for (int kt = 0; kt < 12; ++kt) ay[kt] = load_a(y, 384, kt * 32, lane);
  const v16bf* b1p = (const v16bf*)Wc1p;
  for (int nt = 0; nt < 8; ++nt) {
    v8f acc = zero8();
#pragma unroll
    for (int kt = 0; kt < 12; ++kt)
      acc = wmma_bf16(ay[kt], b1p[(kt * 8 + nt) * 32 + lane], acc);
    const float bias = bc1[nt * 16 + col];
#pragma unroll
    for (int r = 0; r < 8; ++r)
      h[(r + 8 * half) * 128 + nt * 16 + col] = (__bf16)fmaxf(acc[r] + bias, 0.f);
  }
  WAVE_FENCE();

  // h2 = relu(h @ Wc2 + bc2)  (h2 reuses y's LDS space, ld=128)
  __bf16* h2 = y;
  v16bf ah[4];
#pragma unroll
  for (int kt = 0; kt < 4; ++kt) ah[kt] = load_a(h, 128, kt * 32, lane);
  const v16bf* b2p = (const v16bf*)Wc2p;
  for (int nt = 0; nt < 8; ++nt) {
    v8f acc = zero8();
#pragma unroll
    for (int kt = 0; kt < 4; ++kt)
      acc = wmma_bf16(ah[kt], b2p[(kt * 8 + nt) * 32 + lane], acc);
    const float bias = bc2[nt * 16 + col];
#pragma unroll
    for (int r = 0; r < 8; ++r)
      h2[(r + 8 * half) * 128 + nt * 16 + col] = (__bf16)fmaxf(acc[r] + bias, 0.f);
  }
  WAVE_FENCE();

  // out = h2 @ Wc3 + bc3
  v16bf a2[4];
#pragma unroll
  for (int kt = 0; kt < 4; ++kt) a2[kt] = load_a(h2, 128, kt * 32, lane);
  const v16bf* b3p = (const v16bf*)Wc3p;
  for (int nt = 0; nt < 4; ++nt) {
    v8f acc = zero8();
#pragma unroll
    for (int kt = 0; kt < 4; ++kt)
      acc = wmma_bf16(a2[kt], b3p[(kt * 4 + nt) * 32 + lane], acc);
    const float bias = bc3[nt * 16 + col];
#pragma unroll
    for (int r = 0; r < 8; ++r)
      out[(size_t)(row0 + r + 8 * half) * DD + nt * 16 + col] = acc[r] + bias;
  }
}

// ---------------- edge MLP ----------------
__global__ void k4_edge(const float* __restrict__ edge,
                        const float* __restrict__ cyc_out,   // d_out + NE*D
                        const int* __restrict__ c2e,
                        const __bf16* __restrict__ We1p, const float* __restrict__ be1,
                        const __bf16* __restrict__ We2p, const float* __restrict__ be2,
                        float* __restrict__ edge_out) {
  extern __shared__ char smem[];
  const int lane = threadIdx.x & 31;
  const int wid = threadIdx.x >> 5;
  const int tile = blockIdx.x * 4 + wid;
  if (tile >= NEg / 16) return;
  __bf16* ain = (__bf16*)(smem + wid * 8192);
  __bf16* h1 = ain + 16 * 128;

  const int e0 = tile * 16;
  for (int t = lane; t < 16 * DD; t += 32) {
    const int r = t >> 6, c = t & 63;
    ain[r * 128 + c] = (__bf16)edge[(size_t)(e0 + r) * DD + c];
    float s = 0.f;
#pragma unroll
    for (int j = 0; j < 4; ++j) {
      const int idx = c2e[(size_t)(e0 + r) * 4 + j];
      s += cyc_out[(size_t)idx * DD + c];
    }
    ain[r * 128 + DD + c] = (__bf16)s;
  }
  WAVE_FENCE();

  const int col = lane & 15, half = (lane >> 4) & 1;
  v16bf afr[4];
#pragma unroll
  for (int kt = 0; kt < 4; ++kt) afr[kt] = load_a(ain, 128, kt * 32, lane);
  const v16bf* bW1 = (const v16bf*)We1p;
  for (int nt = 0; nt < 8; ++nt) {
    v8f acc = zero8();
#pragma unroll
    for (int kt = 0; kt < 4; ++kt)
      acc = wmma_bf16(afr[kt], bW1[(kt * 8 + nt) * 32 + lane], acc);
    const float bias = be1[nt * 16 + col];
#pragma unroll
    for (int r = 0; r < 8; ++r)
      h1[(r + 8 * half) * 128 + nt * 16 + col] = (__bf16)fmaxf(acc[r] + bias, 0.f);
  }
  WAVE_FENCE();

  v16bf af2[4];
#pragma unroll
  for (int kt = 0; kt < 4; ++kt) af2[kt] = load_a(h1, 128, kt * 32, lane);
  const v16bf* bW2 = (const v16bf*)We2p;
  for (int nt = 0; nt < 4; ++nt) {
    v8f acc = zero8();
#pragma unroll
    for (int kt = 0; kt < 4; ++kt)
      acc = wmma_bf16(af2[kt], bW2[(kt * 4 + nt) * 32 + lane], acc);
    const float bias = be2[nt * 16 + col];
#pragma unroll
    for (int r = 0; r < 8; ++r)
      edge_out[(size_t)(e0 + r + 8 * half) * DD + nt * 16 + col] = acc[r] + bias;  // no relu
  }
}

// ---------------- launcher ----------------
extern "C" void kernel_launch(void* const* d_in, const int* in_sizes, int n_in,
                              void* d_out, int out_size, void* d_ws, size_t ws_size,
                              hipStream_t stream) {
  const float* edge = (const float*)d_in[0];
  const float* cyc5 = (const float*)d_in[1];
  const float* cyc6 = (const float*)d_in[2];
  const int* e2c5 = (const int*)d_in[3];
  const int* e2c6 = (const int*)d_in[4];
  const int* c2e = (const int*)d_in[5];
  const float *W1 = (const float*)d_in[6], *b1 = (const float*)d_in[7];
  const float *W2 = (const float*)d_in[8], *b2 = (const float*)d_in[9];
  const float *Wc1 = (const float*)d_in[10], *bc1 = (const float*)d_in[11];
  const float *Wc2 = (const float*)d_in[12], *bc2 = (const float*)d_in[13];
  const float *Wc3 = (const float*)d_in[14], *bc3 = (const float*)d_in[15];
  const float *We1 = (const float*)d_in[16], *be1 = (const float*)d_in[17];
  const float *We2 = (const float*)d_in[18], *be2 = (const float*)d_in[19];
  const float *Wid5 = (const float*)d_in[20], *Wsum5 = (const float*)d_in[21], *bab5 = (const float*)d_in[22];
  const float *Wid6 = (const float*)d_in[23], *Wsum6 = (const float*)d_in[24], *bab6 = (const float*)d_in[25];

  // workspace layout
  char* ws = (char*)d_ws;
  const size_t XBYTES = (size_t)RTg * DD * sizeof(float);                 // 84,480,000
  const size_t SBYTES = (size_t)(NC5g + NC6g) * DD * sizeof(float);       // 15,360,000
  float* x_all = (float*)ws;
  float* ssum = (float*)(ws + XBYTES);
  __bf16* wp = (__bf16*)(ws + XBYTES + SBYTES);
  __bf16* W1p = wp + 0;
  __bf16* W2p = wp + 16384;
  __bf16* Wc1p = wp + 24576;
  __bf16* Wc2p = wp + 73728;
  __bf16* Wc3p = wp + 90112;
  __bf16* We1p = wp + 98304;
  __bf16* We2p = wp + 114688;
  __bf16* Wid5p = wp + 122880;
  __bf16* Wsum5p = wp + 139264;
  __bf16* Wid6p = wp + 155648;
  __bf16* Wsum6p = wp + 172032;

  struct PackDesc { const float* W; __bf16* out; int K; int N; };
  const PackDesc pd[11] = {
      {W1, W1p, 128, 128},  {W2, W2p, 128, 64},   {Wc1, Wc1p, 384, 128},
      {Wc2, Wc2p, 128, 128},{Wc3, Wc3p, 128, 64}, {We1, We1p, 128, 128},
      {We2, We2p, 128, 64}, {Wid5, Wid5p, 64, 256},{Wsum5, Wsum5p, 64, 256},
      {Wid6, Wid6p, 64, 256},{Wsum6, Wsum6p, 64, 256}};
  for (int i = 0; i < 11; ++i) {
    const int total = (pd[i].K / 32) * (pd[i].N / 16) * 32;
    pack_w<<<(total + 255) / 256, 256, 0, stream>>>(pd[i].W, pd[i].out, pd[i].K, pd[i].N);
  }

  float* out_f = (float*)d_out;
  float* out5 = out_f + (size_t)NEg * DD;
  float* out6 = out5 + (size_t)R5g * DD;

  // stage 1: 330000/16 = 20625 tiles, 4 waves/block
  k1_stage1<<<(RTg / 16 + 3) / 4, 128, 32768, stream>>>(
      edge, cyc5, cyc6, e2c5, e2c6, W1p, b1, W2p, b2, x_all);

  // segment sums: 60000 cycles * 16 float4 lanes
  k2_segsum<<<((NC5g + NC6g) * 16 + 255) / 256, 256, 0, stream>>>(x_all, ssum);

  // stage 2, size 5: 9375 tiles, 2 waves/block
  k3_stage2<<<(R5g / 16 + 1) / 2, 64, 32768, stream>>>(
      x_all, ssum, Wid5p, Wsum5p, bab5, Wc1p, bc1, Wc2p, bc2, Wc3p, bc3,
      out5, R5g, 5, 0, 0);
  // stage 2, size 6: 11250 tiles
  k3_stage2<<<(R6g / 16 + 1) / 2, 64, 32768, stream>>>(
      x_all, ssum, Wid6p, Wsum6p, bab6, Wc1p, bc1, Wc2p, bc2, Wc3p, bc3,
      out6, R6g, 6, R5g, NC5g);

  // edge MLP: 12500 tiles, 4 waves/block; gathers from out5/out6 region of d_out
  k4_edge<<<(NEg / 16 + 3) / 4, 128, 32768, stream>>>(
      edge, out5, c2e, We1p, be1, We2p, be2, out_f);
}
